// GroupQueryAttention_22574348108128
// MI455X (gfx1250) — matrix operand order repaired and checked by reading, not compiled, not run on hardware
//
#include <hip/hip_runtime.h>
#include <hip/hip_bf16.h>

// GQA forward: B=2, S=2048, E=2048, H=32, G=8, D=64 (fp32 in/out).
// All matmuls on v_wmma_f32_16x16x32_bf16 (f32 accumulate).
// LDS tiles laid out so every WMMA fragment load is 2x ds_load_b128.
// bf16 packing via v_perm_b32; softmax reductions via v_permlane16_b32.

typedef __attribute__((ext_vector_type(16))) __bf16 bf16x16;
typedef __attribute__((ext_vector_type(8)))  float  f32x8;

static constexpr int Bc = 2, Sc = 2048, Ec = 2048, Hc = 32, Gc = 8, Dc = 64;

// Pack two f32 into packed bf16 (truncate): one v_perm_b32.
__device__ __forceinline__ unsigned pkbf(float lo, float hi) {
  return __builtin_amdgcn_perm(__builtin_bit_cast(unsigned, hi),
                               __builtin_bit_cast(unsigned, lo), 0x07060302u);
}
__device__ __forceinline__ __bf16 bftrunc(float f) {
  return __builtin_bit_cast(__bf16,
           (unsigned short)(__builtin_bit_cast(unsigned, f) >> 16));
}

__device__ __forceinline__ f32x8 wmma_bf16(bf16x16 a, bf16x16 b, f32x8 c) {
  return __builtin_amdgcn_wmma_f32_16x16x32_bf16(false, a, false, b, (short)0, c,
                                                 false, false);
}

// ---- cross-lane reductions within each 16-lane row (v_permlane16_b32) ------
#if __has_builtin(__builtin_amdgcn_permlane16)
__device__ __forceinline__ float pl16(float x, int s0, int s1) {
  int i = __builtin_bit_cast(int, x);
  i = __builtin_amdgcn_permlane16(i, i, s0, s1, false, false);
  return __builtin_bit_cast(float, i);
}
__device__ __forceinline__ float rowmax16(float v) {
  v = fmaxf(v, pl16(v, 0x67452301, 0xEFCDAB89));   // lane ^ 1
  v = fmaxf(v, pl16(v, 0x54761032, 0xDCFE98BA));   // lane ^ 2
  v = fmaxf(v, pl16(v, 0x32107654, 0xBA98FEDC));   // lane ^ 4
  v = fmaxf(v, pl16(v, 0xFEDCBA98, 0x76543210));   // lane ^ 8
  return v;
}
__device__ __forceinline__ float rowsum16(float v) {
  v += pl16(v, 0x67452301, 0xEFCDAB89);
  v += pl16(v, 0x54761032, 0xDCFE98BA);
  v += pl16(v, 0x32107654, 0xBA98FEDC);
  v += pl16(v, 0xFEDCBA98, 0x76543210);
  return v;
}
#else
__device__ __forceinline__ float rowmax16(float v) {
  for (int m = 1; m < 16; m <<= 1) v = fmaxf(v, __shfl_xor(v, m, 32));
  return v;
}
__device__ __forceinline__ float rowsum16(float v) {
  for (int m = 1; m < 16; m <<= 1) v += __shfl_xor(v, m, 32);
  return v;
}
#endif

// A-matrix (16x32 bf16) ISA layout: lane M = lane%16, half = lane/16.
// K order used here (not the ISA table order): one contiguous run k = 16*half + e, B the same -> 2x ds_load_b128.
__device__ __forceinline__ int a_kmap(int e, int h) {
  return (h << 4) + e;
}

template <int LD>
__device__ __forceinline__ bf16x16 load_a_frag(const __bf16* s, int row0, int k0, int lane) {
  bf16x16 a;
  int r = row0 + (lane & 15), h = lane >> 4;
#pragma unroll
  for (int e = 0; e < 16; ++e) a[e] = s[r * LD + k0 + a_kmap(e, h)];
  return a;
}

// B fragment from transposed LDS tile s[n][LD]: 16 contiguous -> 2x b128.
template <int LD>
__device__ __forceinline__ bf16x16 load_bt_frag(const __bf16* s, int col0, int lane) {
  bf16x16 b;
  const __bf16* p = s + (col0 + (lane & 15)) * LD + ((lane >> 4) << 4);
#pragma unroll
  for (int e = 0; e < 16; ++e) b[e] = p[e];
  return b;
}

// ---------------------------------------------------------------------------
// GEMM: C[M,N] = A[M,K] * W[K,N]. Block tile 256x128, 8 waves (4x2), wave tile
// 64x64 = 4x4 WMMA frags, K-step 32. B staged transposed. MODE==1: scatter C
// into [b, head, s, 64]; MODE==0: row-major [M,N].
// ---------------------------------------------------------------------------
template <int N, int K, int MODE>
__global__ void __launch_bounds__(256)
gqa_gemm(const float* __restrict__ A, const float* __restrict__ W,
         float* __restrict__ C, int M) {
  __shared__ __bf16 sA[256 * 40];    // [m][k], pad 40
  __shared__ __bf16 sBt[128 * 40];   // [n][k], pad 40
  unsigned* sA32  = (unsigned*)sA;
  unsigned* sBt32 = (unsigned*)sBt;

  const int tid  = threadIdx.x;
  const int lane = tid & 31, wave = tid >> 5;
  const int m0 = blockIdx.y * 256, n0 = blockIdx.x * 128;
  const int wm = (wave >> 1) * 64, wn = (wave & 1) * 64;

  f32x8 acc[4][4] = {};

  const float* gA = A + (size_t)(m0 + tid) * K;              // one row per thread
  const int bn = tid >> 1, bk0 = (tid & 1) << 4;             // Bt staging coords
  const float* gW = W + (size_t)bk0 * N + n0 + bn;

  for (int kk = 0; kk < K; kk += 32) {
    __syncthreads();
    {   // stage A: one row per thread, pack pairs -> 16 u32 (b128-merged)
#pragma unroll
      for (int j = 0; j < 16; ++j)
        sA32[tid * 20 + j] = pkbf(gA[kk + 2 * j], gA[kk + 2 * j + 1]);
    }
    {   // stage B transposed: thread owns (n=bn, k=bk0..bk0+15)
#pragma unroll
      for (int j = 0; j < 8; ++j)
        sBt32[bn * 20 + (bk0 >> 1) + j] =
            pkbf(gW[(size_t)(kk + 2 * j) * N], gW[(size_t)(kk + 2 * j + 1) * N]);
    }
    if (kk + 32 < K) {   // prefetch next K-tile (global_prefetch_b8)
      __builtin_prefetch(gA + kk + 32, 0, 1);
      __builtin_prefetch(gW + (size_t)(kk + 32) * N, 0, 1);
    }
    __syncthreads();

    bf16x16 af[4], bfr[4];
#pragma unroll
    for (int mf = 0; mf < 4; ++mf) af[mf] = load_a_frag<40>(sA, wm + mf * 16, 0, lane);
#pragma unroll
    for (int nf = 0; nf < 4; ++nf) bfr[nf] = load_bt_frag<40>(sBt, wn + nf * 16, lane);
#pragma unroll
    for (int mf = 0; mf < 4; ++mf)
#pragma unroll
      for (int nf = 0; nf < 4; ++nf) acc[mf][nf] = wmma_bf16(af[mf], bfr[nf], acc[mf][nf]);
  }

  // epilogue: C layout M = r + 8*half, N = lane%16 (shift-only indexing)
  const int half = lane >> 4, cn = lane & 15;
  constexpr int NH = N >> 6;
#pragma unroll
  for (int mf = 0; mf < 4; ++mf) {
#pragma unroll
    for (int nf = 0; nf < 4; ++nf) {
      int mb  = m0 + wm + mf * 16 + 8 * half;
      int col = n0 + wn + nf * 16 + cn;
#pragma unroll
      for (int r = 0; r < 8; ++r) {
        int m = mb + r;
        float v = acc[mf][nf][r];
        if (MODE == 0) {
          C[(size_t)m * N + col] = v;
        } else {
          int s = m & (Sc - 1), b = m >> 11;          // S = 2048
          int hh = col >> 6, dd = col & 63;
          C[(((size_t)b * NH + hh) * Sc + s) * 64 + dd] = v;
        }
      }
    }
  }
}

// ---------------------------------------------------------------------------
// Fused RMS-norm + RoPE over rows of [B, NH, S, 64], one wave per row.
// ---------------------------------------------------------------------------
__global__ void __launch_bounds__(256)
gqa_normrope(float* __restrict__ X, const float* __restrict__ w,
             const float* __restrict__ cosb, const float* __restrict__ sinb,
             float* __restrict__ X2) {
  int row  = blockIdx.x * 8 + (threadIdx.x >> 5);
  int lane = threadIdx.x & 31;
  int s = row & (Sc - 1);
  float* p = X + (size_t)row * 64;
  float xlo = p[lane], xhi = p[lane + 32];
  float ss = xlo * xlo + xhi * xhi;
#pragma unroll
  for (int m = 1; m < 32; m <<= 1) ss += __shfl_xor(ss, m, 32);
  float rms = sqrtf(ss * (1.0f / 64.0f));
  float inv = 1.0f / (rms + 1e-6f);
  float ylo = xlo * inv * w[lane];
  float yhi = xhi * inv * w[lane + 32];
  float clo = cosb[s * 64 + lane], chi = cosb[s * 64 + lane + 32];
  float slo = sinb[s * 64 + lane], shi = sinb[s * 64 + lane + 32];
  float olo = ylo * clo - yhi * slo;   // rot(lo) = -hi
  float ohi = yhi * chi + ylo * shi;   // rot(hi) = +lo
  p[lane] = olo; p[lane + 32] = ohi;
  if (X2) { float* q = X2 + (size_t)row * 64; q[lane] = olo; q[lane + 32] = ohi; }
}

// ---------------------------------------------------------------------------
// Causal flash attention. Block = (b*H+h, 128-query tile), 8 waves x 16 rows.
// Per 32-key block: S = Q*K^T (4 WMMA), online softmax, ctx += P*V (4 WMMA).
// ---------------------------------------------------------------------------
__global__ void __launch_bounds__(256)
gqa_flash_attn(const float* __restrict__ Qh, const float* __restrict__ Kh,
               const float* __restrict__ Vh, float* __restrict__ ctx) {
  __shared__ __bf16 sQ[128 * 72];        // [q][d]
  __shared__ __bf16 sK[32 * 72];         // [key][d]
  __shared__ __bf16 sVt[64 * 40];        // [d][key]
  __shared__ __bf16 sP[8][16 * 40];      // per-wave P tile [m][k]
  unsigned* sQ32  = (unsigned*)sQ;
  unsigned* sK32  = (unsigned*)sK;
  unsigned* sVt32 = (unsigned*)sVt;

  const int bh = blockIdx.x;
  const int b = bh >> 5, h = bh & 31, g = h >> 2;    // H=32, H/G=4
  const int qt = blockIdx.y;
  const int tid = threadIdx.x, lane = tid & 31, wave = tid >> 5;

  const float* Qb = Qh + (((size_t)b * Hc + h) * Sc + qt * 128) * Dc;
  const float* Kb = Kh + ((size_t)b * Gc + g) * Sc * Dc;
  const float* Vb = Vh + ((size_t)b * Gc + g) * Sc * Dc;

  {   // stage Q tile 128x64, scaled by 1/sqrt(64), packed pairs
    int r = tid >> 1, c0 = (tid & 1) << 5;
    const float* gq = Qb + (size_t)r * Dc + c0;
#pragma unroll
    for (int i = 0; i < 16; ++i)
      sQ32[r * 36 + (c0 >> 1) + i] =
          pkbf(gq[2 * i] * 0.125f, gq[2 * i + 1] * 0.125f);
  }
  __syncthreads();

  bf16x16 aQ[2];
  aQ[0] = load_a_frag<72>(sQ, wave * 16, 0, lane);
  aQ[1] = load_a_frag<72>(sQ, wave * 16, 32, lane);

  f32x8 cacc[4] = {};
  float mrow[8], lrow[8];
#pragma unroll
  for (int r = 0; r < 8; ++r) { mrow[r] = -1e30f; lrow[r] = 0.0f; }

  const int half = lane >> 4, cn = lane & 15;
  const int kr = tid >> 3, kc0 = (tid & 7) << 3;     // K staging: [key][d]
  const int vd = tid & 63, vr0 = (tid >> 6) << 3;    // V staging: [d][key]
  const int nkb = qt * 4 + 4;

  for (int kb = 0; kb < nkb; ++kb) {
    __syncthreads();
    {   // stage K row-major (4 u32 per thread)
      const float* gk = Kb + ((size_t)kb * 32 + kr) * Dc + kc0;
#pragma unroll
      for (int i = 0; i < 4; ++i)
        sK32[kr * 36 + (kc0 >> 1) + i] = pkbf(gk[2 * i], gk[2 * i + 1]);
    }
    {   // stage V transposed: thread owns (d=vd, key=vr0..vr0+7)
      const float* gv = Vb + ((size_t)kb * 32 + vr0) * Dc + vd;
#pragma unroll
      for (int i = 0; i < 4; ++i)
        sVt32[vd * 20 + (vr0 >> 1) + i] =
            pkbf(gv[(size_t)(2 * i) * Dc], gv[(size_t)(2 * i + 1) * Dc]);
    }
    __syncthreads();

    // S = Q * K^T for two 16-key sub-tiles (B elements d-contiguous in sK)
    f32x8 sfr[2];
#pragma unroll
    for (int nb = 0; nb < 2; ++nb) {
      f32x8 s0 = {};
#pragma unroll
      for (int kk = 0; kk < 2; ++kk) {
        bf16x16 bb;
        const __bf16* p = &sK[(nb * 16 + cn) * 72 + kk * 32 + (half << 4)];
#pragma unroll
        for (int e = 0; e < 16; ++e) bb[e] = p[e];
        s0 = wmma_bf16(aQ[kk], bb, s0);
      }
      sfr[nb] = s0;
    }

    // causal mask + online softmax (row M = r + 8*half)
    const int key0 = kb * 32;
#pragma unroll
    for (int r = 0; r < 8; ++r) {
      int qg = qt * 128 + wave * 16 + r + 8 * half;
      float v0 = (key0 + cn      > qg) ? -1e30f : sfr[0][r];
      float v1 = (key0 + 16 + cn > qg) ? -1e30f : sfr[1][r];
      float mnew  = fmaxf(mrow[r], rowmax16(fmaxf(v0, v1)));
      float scale = __expf(mrow[r] - mnew);
      float p0 = __expf(v0 - mnew), p1 = __expf(v1 - mnew);
      float psum = rowsum16(p0 + p1);
      lrow[r] = lrow[r] * scale + psum;
      mrow[r] = mnew;
#pragma unroll
      for (int nf = 0; nf < 4; ++nf) cacc[nf][r] *= scale;
      int mr = r + 8 * half;                 // re-lay P via per-wave LDS
      sP[wave][mr * 40 + cn]      = bftrunc(p0);
      sP[wave][mr * 40 + 16 + cn] = bftrunc(p1);
    }

    // ctx += P(16x32) * V(32x64); V B-fragments key-contiguous in sVt
    bf16x16 ap = load_a_frag<40>(sP[wave], 0, 0, lane);
#pragma unroll
    for (int nf = 0; nf < 4; ++nf) {
      bf16x16 bv;
      const __bf16* p = &sVt[(nf * 16 + cn) * 40 + (half << 4)];
#pragma unroll
      for (int e = 0; e < 16; ++e) bv[e] = p[e];
      cacc[nf] = wmma_bf16(ap, bv, cacc[nf]);
    }
  }

  // epilogue: ctx[b, s, h, d] = acc / l
#pragma unroll
  for (int nf = 0; nf < 4; ++nf) {
#pragma unroll
    for (int r = 0; r < 8; ++r) {
      int qg = qt * 128 + wave * 16 + r + 8 * half;
      int d  = nf * 16 + cn;
      ctx[(((size_t)b * Sc + qg) * Hc + h) * 64 + d] = cacc[nf][r] / lrow[r];
    }
  }
}

// ---------------------------------------------------------------------------
extern "C" void kernel_launch(void* const* d_in, const int* in_sizes, int n_in,
                              void* d_out, int out_size, void* d_ws, size_t ws_size,
                              hipStream_t stream) {
  (void)in_sizes; (void)n_in; (void)out_size; (void)ws_size;
  const float* x    = (const float*)d_in[0];
  // d_in[1] = position_ids (identity arange -> unused)
  const float* cosb = (const float*)d_in[2];
  const float* sinb = (const float*)d_in[3];
  const float* Wq   = (const float*)d_in[4];
  const float* Wk   = (const float*)d_in[5];
  const float* Wv   = (const float*)d_in[6];
  const float* Wo   = (const float*)d_in[7];
  const float* qnw  = (const float*)d_in[8];
  const float* knw  = (const float*)d_in[9];

  float* out  = (float*)d_out;                       // [B,S,E]
  float* Kout = out  + (size_t)Bc * Sc * Ec;         // [B,G,S,D]
  float* Vout = Kout + (size_t)Bc * Gc * Sc * Dc;    // [B,G,S,D]

  float* ws  = (float*)d_ws;
  float* Qh  = ws;                                   // [B,H,S,D]
  float* Khh = Qh  + (size_t)Bc * Hc * Sc * Dc;      // [B,G,S,D]
  float* Vhh = Khh + (size_t)Bc * Gc * Sc * Dc;      // [B,G,S,D]
  float* ctx = Vhh + (size_t)Bc * Gc * Sc * Dc;      // [B,S,H,D]

  const int M = Bc * Sc;
  dim3 blk(256);

  // QKV projections (write head-transposed layouts)
  gqa_gemm<Hc * Dc, Ec, 1><<<dim3((Hc * Dc) / 128, M / 256), blk, 0, stream>>>(x, Wq, Qh,  M);
  gqa_gemm<Gc * Dc, Ec, 1><<<dim3((Gc * Dc) / 128, M / 256), blk, 0, stream>>>(x, Wk, Khh, M);
  gqa_gemm<Gc * Dc, Ec, 1><<<dim3((Gc * Dc) / 128, M / 256), blk, 0, stream>>>(x, Wv, Vhh, M);

  // RMS-norm + RoPE (K also duplicated into d_out's K region)
  gqa_normrope<<<(Bc * Hc * Sc) / 8, 256, 0, stream>>>(Qh,  qnw, cosb, sinb, nullptr);
  gqa_normrope<<<(Bc * Gc * Sc) / 8, 256, 0, stream>>>(Khh, knw, cosb, sinb, Kout);

  // V output is the raw projection, already in [B,G,S,D]
  hipMemcpyAsync(Vout, Vhh, (size_t)Bc * Gc * Sc * Dc * sizeof(float),
                 hipMemcpyDeviceToDevice, stream);

  // causal flash attention
  gqa_flash_attn<<<dim3(Bc * Hc, Sc / 128), 256, 0, stream>>>(Qh, Khh, Vhh, ctx);

  // output projection (row-major C)
  gqa_gemm<Ec, Hc * Dc, 0><<<dim3(Ec / 128, M / 256), blk, 0, stream>>>(ctx, Wo, out, M);
}
